// BitNetMoEFFN_87471303951029
// MI455X (gfx1250) — 8-bit matrix operand order repaired and checked by reading, compiled, not run on hardware
//
#include <hip/hip_runtime.h>

static constexpr int   Bc = 2, Sc = 2048;
static constexpr int   T  = Bc * Sc;
static constexpr int   H  = 2048;
static constexpr int   I  = 4096;
static constexpr int   E  = 8;
static constexpr float EPSv = 1e-5f;

typedef __attribute__((ext_vector_type(8))) int v8i;

static constexpr size_t PERMAT   = (size_t)I * H;
static constexpr size_t O_WQG    = 0;
static constexpr size_t O_WQU    = O_WQG + (size_t)E * PERMAT;
static constexpr size_t O_WQD    = O_WQU + (size_t)E * PERMAT;
static constexpr size_t O_XQ     = O_WQD + (size_t)E * PERMAT;
static constexpr size_t O_ACTQ   = O_XQ  + (size_t)T * H;
static constexpr size_t O_ACT    = O_ACTQ + (size_t)T * I;
static constexpr size_t O_AS     = O_ACT + (size_t)T * I * 4;
static constexpr size_t O_A2S    = O_AS  + (size_t)T * 4;
static constexpr size_t O_COMB   = O_A2S + (size_t)T * 4;
static constexpr size_t O_SCALES = O_COMB + (size_t)T * E * 4;

__global__ void __launch_bounds__(256) k_zero_f32(float* p, size_t n) {
  size_t i = (size_t)blockIdx.x * blockDim.x + threadIdx.x;
  size_t s = (size_t)gridDim.x * blockDim.x;
  for (; i < n; i += s) p[i] = 0.0f;
}

__global__ void __launch_bounds__(256) k_absum(const float* __restrict__ w,
                                               float* __restrict__ sums, int sbase) {
  __shared__ float red[256];
  const size_t chunk4 = PERMAT / (gridDim.x * 4);
  const float4* base = (const float4*)(w + (size_t)blockIdx.y * PERMAT) + (size_t)blockIdx.x * chunk4;
  float acc = 0.0f;
  for (size_t i = threadIdx.x; i < chunk4; i += 256) {
    float4 v = base[i];
    acc += fabsf(v.x) + fabsf(v.y) + fabsf(v.z) + fabsf(v.w);
  }
  red[threadIdx.x] = acc;
  __syncthreads();
  for (int s = 128; s > 0; s >>= 1) {
    if (threadIdx.x < s) red[threadIdx.x] += red[threadIdx.x + s];
    __syncthreads();
  }
  if (threadIdx.x == 0) atomicAdd(&sums[sbase + blockIdx.y], red[0]);
}

__global__ void k_finalize_scales(float* sc) {
  int i = threadIdx.x;
  if (i < 24) {
    float mean = sc[i] / (float)PERMAT;
    float dq = fmaxf(mean, EPSv);
    sc[32 + i] = 1.0f / dq;
    sc[64 + i] = dq;
  }
}

__device__ __forceinline__ signed char tern_q(float v, float s) {
  float r = rintf(v * s);
  return (signed char)fminf(fmaxf(r, -1.0f), 1.0f);
}

__global__ void __launch_bounds__(256) k_quantw(const float* __restrict__ w,
                                                signed char* __restrict__ q,
                                                const float* __restrict__ sc, int sbase,
                                                size_t total4) {
  const float4* w4 = (const float4*)w;
  char4* q4 = (char4*)q;
  size_t i = (size_t)blockIdx.x * blockDim.x + threadIdx.x;
  size_t st = (size_t)gridDim.x * blockDim.x;
  for (; i < total4; i += st) {
    int m = (int)((i * 4) / PERMAT);
    float s = sc[32 + sbase + m];
    float4 v = w4[i];
    char4 o;
    o.x = tern_q(v.x, s); o.y = tern_q(v.y, s);
    o.z = tern_q(v.z, s); o.w = tern_q(v.w, s);
    q4[i] = o;
  }
}

__device__ __forceinline__ signed char i8_q(float v, float s) {
  float r = rintf(v * s);
  return (signed char)fminf(fmaxf(r, -128.0f), 127.0f);
}

__global__ void __launch_bounds__(256) k_quant_rows(const float* __restrict__ src,
                                                    signed char* __restrict__ dst,
                                                    float* __restrict__ row_s, int K) {
  __shared__ float red[256];
  const size_t t = blockIdx.x;
  const float4* s4 = (const float4*)(src + t * K);
  char4* d4 = (char4*)(dst + t * K);
  const int K4 = K >> 2;
  float m = 0.0f;
  for (int k = threadIdx.x; k < K4; k += 256) {
    float4 v = s4[k];
    m = fmaxf(m, fmaxf(fmaxf(fabsf(v.x), fabsf(v.y)), fmaxf(fabsf(v.z), fabsf(v.w))));
  }
  red[threadIdx.x] = m;
  __syncthreads();
  for (int s = 128; s > 0; s >>= 1) {
    if (threadIdx.x < s) red[threadIdx.x] = fmaxf(red[threadIdx.x], red[threadIdx.x + s]);
    __syncthreads();
  }
  const float am = fmaxf(red[0], EPSv);
  const float scq = 127.0f / am;
  for (int k = threadIdx.x; k < K4; k += 256) {
    float4 v = s4[k];
    char4 o;
    o.x = i8_q(v.x, scq); o.y = i8_q(v.y, scq);
    o.z = i8_q(v.z, scq); o.w = i8_q(v.w, scq);
    d4[k] = o;
  }
  if (threadIdx.x == 0) row_s[t] = am / 127.0f;
}

__global__ void __launch_bounds__(256) k_router(const float* __restrict__ x,
                                                const float* __restrict__ rw,
                                                float* __restrict__ combine) {
  __shared__ float red[E][256];
  const size_t t = blockIdx.x;
  float p[E];
#pragma unroll
  for (int e = 0; e < E; ++e) p[e] = 0.0f;
  for (int h = threadIdx.x; h < H; h += 256) {
    float xv = x[t * H + h];
#pragma unroll
    for (int e = 0; e < E; ++e) p[e] += xv * rw[e * H + h];
  }
#pragma unroll
  for (int e = 0; e < E; ++e) red[e][threadIdx.x] = p[e];
  __syncthreads();
  for (int s = 128; s > 0; s >>= 1) {
    if (threadIdx.x < s)
#pragma unroll
      for (int e = 0; e < E; ++e) red[e][threadIdx.x] += red[e][threadIdx.x + s];
    __syncthreads();
  }
  if (threadIdx.x == 0) {
    float lg[E], mx = -3.4e38f;
#pragma unroll
    for (int e = 0; e < E; ++e) { lg[e] = red[e][0]; mx = fmaxf(mx, lg[e]); }
    float pe[E], sum = 0.0f;
#pragma unroll
    for (int e = 0; e < E; ++e) { pe[e] = expf(lg[e] - mx); sum += pe[e]; }
#pragma unroll
    for (int e = 0; e < E; ++e) pe[e] /= sum;
    int i1 = 0;
#pragma unroll
    for (int e = 1; e < E; ++e) if (pe[e] > pe[i1]) i1 = e;
    int i2 = (i1 == 0) ? 1 : 0;
#pragma unroll
    for (int e = 0; e < E; ++e) if (e != i1 && pe[e] > pe[i2]) i2 = e;
    float wsum = pe[i1] + pe[i2];
#pragma unroll
    for (int e = 0; e < E; ++e) combine[t * E + e] = 0.0f;
    combine[t * E + i1] = pe[i1] / wsum;
    combine[t * E + i2] = pe[i2] / wsum;
  }
}

__device__ __forceinline__ v8i frag_ld(const char* p) {
  return *(const v8i*)p;
}

__global__ void __launch_bounds__(256) k_gateup(const signed char* __restrict__ xq,
                                                const signed char* __restrict__ wgq,
                                                const signed char* __restrict__ wuq,
                                                const float* __restrict__ a_s,
                                                const float* __restrict__ sc, int e,
                                                float* __restrict__ act) {
  __shared__ __align__(32) char sA[2][64 * 64];
  __shared__ __align__(32) char sBg[2][64 * 64];
  __shared__ __align__(32) char sBu[2][64 * 64];
  const int n0 = blockIdx.x * 64;
  const int t0 = blockIdx.y * 64;
  const int tid = threadIdx.x, lane = tid & 31, wave = tid >> 5;
  const int mtile = (wave & 3) * 16;
  const int ntile = (wave >> 2) * 32;
  const signed char* g = wgq + (size_t)e * PERMAT;
  const signed char* u = wuq + (size_t)e * PERMAT;
  const int row = tid >> 2, s4 = tid & 3;
  const size_t gA = (size_t)(t0 + row) * H + s4 * 16;
  const size_t gB = (size_t)(n0 + row) * H + s4 * 16;
  const int aoff = row * 64 + s4 * 8;
  const int boff = row * 64 + ((s4 & 1) << 5) + ((s4 >> 1) << 4);

  const int lrow = lane & 15, half = lane >> 4;
  const int afr = (mtile + lrow) * 64 + half * 32;
  const int bfr = lrow * 64 + half * 32;

  v8i accg0 = {0,0,0,0,0,0,0,0}, accg1 = accg0, accu0 = accg0, accu1 = accg0;

  auto stage = [&](int k0, int buf) {
    int4 va = *(const int4*)(xq + gA + k0);
    int4 vg = *(const int4*)(g + gB + k0);
    int4 vu = *(const int4*)(u + gB + k0);
    *(int4*)(sA[buf] + boff) = va;
    *(int4*)(sBg[buf] + boff) = vg;
    *(int4*)(sBu[buf] + boff) = vu;
  };

  stage(0, 0);
  const int nk = H / 64;
  for (int k = 0; k < nk; ++k) {
    __syncthreads();
    const int cur = k & 1;
    if (k + 1 < nk) stage((k + 1) * 64, cur ^ 1);
    if (k + 2 < nk) {
      __builtin_prefetch(xq + gA + (size_t)(k + 2) * 64, 0, 3);
      __builtin_prefetch(g + gB + (size_t)(k + 2) * 64, 0, 3);
      __builtin_prefetch(u + gB + (size_t)(k + 2) * 64, 0, 3);
    }
    v8i a   = frag_ld(sA[cur] + afr);
    v8i bg0 = frag_ld(sBg[cur] + ntile * 64 + bfr);
    v8i bg1 = frag_ld(sBg[cur] + (ntile + 16) * 64 + bfr);
    v8i bu0 = frag_ld(sBu[cur] + ntile * 64 + bfr);
    v8i bu1 = frag_ld(sBu[cur] + (ntile + 16) * 64 + bfr);
    accg0 = __builtin_amdgcn_wmma_i32_16x16x64_iu8(true, a, true, bg0, accg0, false, false);
    accg1 = __builtin_amdgcn_wmma_i32_16x16x64_iu8(true, a, true, bg1, accg1, false, false);
    accu0 = __builtin_amdgcn_wmma_i32_16x16x64_iu8(true, a, true, bu0, accu0, false, false);
    accu1 = __builtin_amdgcn_wmma_i32_16x16x64_iu8(true, a, true, bu1, accu1, false, false);
  }

  const float gs = sc[64 + e];
  const float us = sc[64 + 8 + e];
#pragma unroll
  for (int sub = 0; sub < 2; ++sub) {
#pragma unroll
    for (int r = 0; r < 8; ++r) {
      const int t = t0 + mtile + half * 8 + r;
      const int n = n0 + ntile + sub * 16 + lrow;
      const float as = a_s[t];
      const int gi = sub ? accg1[r] : accg0[r];
      const int ui = sub ? accu1[r] : accu0[r];
      const float gv = (float)gi * as * gs;
      const float uv = (float)ui * as * us;
      const float rg = gv > 0.0f ? gv : 0.0f;
      act[(size_t)t * I + n] = uv * rg * rg;
    }
  }
}

__global__ void __launch_bounds__(256) k_down(const signed char* __restrict__ actq,
                                              const signed char* __restrict__ wdw,
                                              const float* __restrict__ a2s,
                                              const float* __restrict__ sc,
                                              const float* __restrict__ combine, int e,
                                              float* __restrict__ out) {
  __shared__ __align__(32) char sA[2][64 * 64];
  __shared__ __align__(32) char sB[2][64 * 64];
  const int n0 = blockIdx.x * 64;
  const int t0 = blockIdx.y * 64;
  const int tid = threadIdx.x, lane = tid & 31, wave = tid >> 5;
  const int mtile = (wave & 3) * 16;
  const int ntile = (wave >> 2) * 32;
  const signed char* w = wdw + (size_t)e * PERMAT;
  const int row = tid >> 2, s4 = tid & 3;
  const size_t gA = (size_t)(t0 + row) * I + s4 * 16;
  const size_t gB = (size_t)(n0 + row) * I + s4 * 16;
  const int aoff = row * 64 + s4 * 8;
  const int boff = row * 64 + ((s4 & 1) << 5) + ((s4 >> 1) << 4);

  const int lrow = lane & 15, half = lane >> 4;
  const int afr = (mtile + lrow) * 64 + half * 32;
  const int bfr = lrow * 64 + half * 32;

  v8i acc0 = {0,0,0,0,0,0,0,0}, acc1 = acc0;

  auto stage = [&](int k0, int buf) {
    int4 va = *(const int4*)(actq + gA + k0);
    int4 vb = *(const int4*)(w + gB + k0);
    *(int4*)(sA[buf] + boff) = va;
    *(int4*)(sB[buf] + boff) = vb;
  };

  stage(0, 0);
  const int nk = I / 64;
  for (int k = 0; k < nk; ++k) {
    __syncthreads();
    const int cur = k & 1;
    if (k + 1 < nk) stage((k + 1) * 64, cur ^ 1);
    if (k + 2 < nk) {
      __builtin_prefetch(actq + gA + (size_t)(k + 2) * 64, 0, 3);
      __builtin_prefetch(w + gB + (size_t)(k + 2) * 64, 0, 3);
    }
    v8i a  = frag_ld(sA[cur] + afr);
    v8i b0 = frag_ld(sB[cur] + ntile * 64 + bfr);
    v8i b1 = frag_ld(sB[cur] + (ntile + 16) * 64 + bfr);
    acc0 = __builtin_amdgcn_wmma_i32_16x16x64_iu8(true, a, true, b0, acc0, false, false);
    acc1 = __builtin_amdgcn_wmma_i32_16x16x64_iu8(true, a, true, b1, acc1, false, false);
  }

  const float ds = sc[64 + 16 + e];
#pragma unroll
  for (int sub = 0; sub < 2; ++sub) {
#pragma unroll
    for (int r = 0; r < 8; ++r) {
      const int t = t0 + mtile + half * 8 + r;
      const int n = n0 + ntile + sub * 16 + lrow;
      const int yi = sub ? acc1[r] : acc0[r];
      const float y = (float)yi * a2s[t] * ds;
      out[(size_t)t * H + n] += combine[(size_t)t * E + e] * y;
    }
  }
}

extern "C" void kernel_launch(void* const* d_in, const int* in_sizes, int n_in,
                              void* d_out, int out_size, void* d_ws, size_t ws_size,
                              hipStream_t stream) {
  (void)in_sizes; (void)n_in; (void)out_size; (void)ws_size;
  const float* x  = (const float*)d_in[0];
  const float* rw = (const float*)d_in[1];
  const float* gw = (const float*)d_in[2];
  const float* uw = (const float*)d_in[3];
  const float* dw = (const float*)d_in[4];
  float* out = (float*)d_out;
  char*  ws  = (char*)d_ws;

  signed char* wqg  = (signed char*)(ws + O_WQG);
  signed char* wqu  = (signed char*)(ws + O_WQU);
  signed char* wqd  = (signed char*)(ws + O_WQD);
  signed char* xq   = (signed char*)(ws + O_XQ);
  signed char* actq = (signed char*)(ws + O_ACTQ);
  float* act  = (float*)(ws + O_ACT);
  float* a_s  = (float*)(ws + O_AS);
  float* a2s  = (float*)(ws + O_A2S);
  float* comb = (float*)(ws + O_COMB);
  float* sc   = (float*)(ws + O_SCALES);

  k_zero_f32<<<1024, 256, 0, stream>>>(out, (size_t)T * H);
  k_zero_f32<<<1, 256, 0, stream>>>(sc, 96);

  k_absum<<<dim3(512, E), 256, 0, stream>>>(gw, sc, 0);
  k_absum<<<dim3(512, E), 256, 0, stream>>>(uw, sc, 8);
  k_absum<<<dim3(512, E), 256, 0, stream>>>(dw, sc, 16);
  k_finalize_scales<<<1, 32, 0, stream>>>(sc);

  k_quantw<<<8192, 256, 0, stream>>>(gw, wqg, sc, 0,  (size_t)E * PERMAT / 4);
  k_quantw<<<8192, 256, 0, stream>>>(uw, wqu, sc, 8,  (size_t)E * PERMAT / 4);
  k_quantw<<<8192, 256, 0, stream>>>(dw, wqd, sc, 16, (size_t)E * PERMAT / 4);

  k_quant_rows<<<T, 256, 0, stream>>>(x, xq, a_s, H);
  k_router<<<T, 256, 0, stream>>>(x, rw, comb);

  for (int e = 0; e < E; ++e) {
    k_gateup<<<dim3(I / 64, T / 64), 256, 0, stream>>>(xq, wqg, wqu, a_s, sc, e, act);
    k_quant_rows<<<T, 256, 0, stream>>>(act, actq, a2s, I);
    k_down<<<dim3(H / 64, T / 64), 256, 0, stream>>>(actq, wqd, a2s, sc, comb, e, out);
  }
}
